// CausalBoostNet_28973849379275
// MI455X (gfx1250) — compile-verified, matrix operand order checked by reading, not run on hardware
//
#include <hip/hip_runtime.h>
#include <hip/hip_bf16.h>
#include <math.h>

// ---------------------------------------------------------------------------
// CausalBoostNet fused CDNA5 implementation
// B=4096, D=256, H=64, K=32, R=2, C=10.  bf16 WMMA (16x16x32), fp32 accum.
// One workgroup per batch keeps h [256x64] resident in LDS for all rounds;
// weights live in L2 (192 MB); only x, the final-h spill and logits touch HBM.
// ---------------------------------------------------------------------------

#define Bb 4096
#define Dd 256
#define Hh 64
#define Kk 32
#define Rr 2
#define Cc 10
#define FLAT (Dd * Hh)          // 16384
#define H2 (2 * Hh)             // 128

typedef __attribute__((ext_vector_type(16))) __bf16 v16bf;
typedef __attribute__((ext_vector_type(8)))  float  v8f;

union Frag16 { v16bf v; int4 q[2]; };

__device__ __forceinline__ v16bf ld_frag(const __bf16* rowptr, int koff) {
  // A-style fragment load: lane holds row (lane&15); k-chunks at koff and koff+16
  Frag16 f;
  f.q[0] = *(const int4*)(rowptr + koff);
  f.q[1] = *(const int4*)(rowptr + koff + 16);
  return f.v;
}

__device__ __forceinline__ v8f wmma_bf16(v16bf a, v16bf b, v8f c) {
  return __builtin_amdgcn_wmma_f32_16x16x32_bf16(
      /*neg_a=*/false, a, /*neg_b=*/false, b,
      /*c_mod=*/(short)0, c, /*reuse_a=*/false, /*reuse_b=*/false);
}

// C fragment stored TRANSPOSED: lane's 8 rows (M = mbase..mbase+7) are
// contiguous along the destination's fast axis -> single ds_store_b128.
__device__ __forceinline__ void st_transposed(__bf16* dst, v8f c, float bias) {
  union { __bf16 h[8]; int4 q; } o;
#pragma unroll
  for (int v = 0; v < 8; ++v) o.h[v] = (__bf16)(c[v] + bias);
  *(int4*)dst = o.q;
}

// C fragment stored row-major [..][ldn]: 8 strided b16 stores.
__device__ __forceinline__ void st_rowmajor(__bf16* dst, int ldn, v8f c,
                                            float bias, bool do_relu) {
#pragma unroll
  for (int v = 0; v < 8; ++v) {
    float f = c[v] + bias;
    if (do_relu) f = fmaxf(f, 0.0f);
    dst[v * ldn] = (__bf16)f;
  }
}

// ---------------------------------------------------------------------------
// Kernel 1: DAG weight matrix  WT[j][i] = W[i][j]  (bf16), one block.
// ---------------------------------------------------------------------------
__global__ __launch_bounds__(256) void k_wgraph(const float* __restrict__ A_raw,
                                                const float* __restrict__ prio,
                                                const float* __restrict__ el,
                                                __bf16* __restrict__ WT) {
  __shared__ float S[Kk][Kk];    // 4 KB
  __shared__ float T[Dd][Kk];    // 32 KB  (T = A @ S)
  const int t = threadIdx.x;     // 0..255 ; one D-row per thread

  // softmax over A_raw row t  (kept in registers)
  float a[Kk];
  float mx = -1e30f;
#pragma unroll
  for (int k = 0; k < Kk; ++k) { a[k] = A_raw[t * Kk + k]; mx = fmaxf(mx, a[k]); }
  float s = 0.0f;
#pragma unroll
  for (int k = 0; k < Kk; ++k) { a[k] = __expf(a[k] - mx); s += a[k]; }
  const float inv = 1.0f / s;
#pragma unroll
  for (int k = 0; k < Kk; ++k) a[k] *= inv;

  // S = hard_edges * soft_direction * (1 - I)
  for (int idx = t; idx < Kk * Kk; idx += 256) {
    const int i = idx >> 5, j = idx & 31;
    const float dir = 1.0f / (1.0f + __expf(-(prio[j] - prio[i]) * (1.0f / 0.3f)));
    const float edge = (el[i * Kk + j] > 0.0f) ? 1.0f : 0.0f;  // sigmoid(x)>0.5
    S[i][j] = (i == j) ? 0.0f : edge * dir;
  }
  __syncthreads();

  // T[t][k] = sum_m a[m] * S[m][k]
  float trow[Kk];
#pragma unroll
  for (int k = 0; k < Kk; ++k) trow[k] = 0.0f;
#pragma unroll 4
  for (int m = 0; m < Kk; ++m) {
    const float am = a[m];
#pragma unroll
    for (int k = 0; k < Kk; ++k) trow[k] += am * S[m][k];
  }
#pragma unroll
  for (int k = 0; k < Kk; ++k) T[t][k] = trow[k];
  __syncthreads();

  // WT[t][i] = W[i][t] = dot(T[i], A[t]) , zero diagonal
  for (int i = 0; i < Dd; ++i) {
    float acc = 0.0f;
#pragma unroll 8
    for (int k = 0; k < Kk; ++k) acc += T[i][k] * a[k];
    WT[t * Dd + i] = (__bf16)((i == t) ? 0.0f : acc);
  }
}

// ---------------------------------------------------------------------------
// Elementwise fp32 -> bf16 convert / transpose
// ---------------------------------------------------------------------------
__global__ void k_cvt(const float* __restrict__ src, __bf16* __restrict__ dst, int n) {
  for (int i = blockIdx.x * blockDim.x + threadIdx.x; i < n; i += gridDim.x * blockDim.x)
    dst[i] = (__bf16)src[i];
}

// src [rows][cols] row-major -> dst [cols][rows] row-major (bf16)
__global__ void k_tr(const float* __restrict__ src, __bf16* __restrict__ dst,
                     int rows, int cols) {
  const int n = rows * cols;
  for (int i = blockIdx.x * blockDim.x + threadIdx.x; i < n; i += gridDim.x * blockDim.x) {
    const int r = i / cols, c = i - r * cols;
    dst[c * rows + r] = (__bf16)src[i];
  }
}

// ---------------------------------------------------------------------------
// Kernel 2: fused per-batch network.  1 block = 1 batch, 256 threads = 8 waves.
// LDS: hS [256][64] | msgT [64][256] (reused as uS [256][64]) | aggS [256][64]
// ---------------------------------------------------------------------------
__global__ __launch_bounds__(256) void k_main(
    const float* __restrict__ x,
    const __bf16* __restrict__ wn_bf, const __bf16* __restrict__ bn_bf,
    const __bf16* __restrict__ WT,
    const __bf16* __restrict__ aggwT, const float* __restrict__ agg_b,
    const __bf16* __restrict__ w1T,   const float* __restrict__ b1,
    const __bf16* __restrict__ w2T,   const float* __restrict__ b2,
    __bf16* __restrict__ hflat) {
  extern __shared__ __bf16 sm[];
  __bf16* hS   = sm;                  // [256][64]
  __bf16* msgT = sm + Dd * Hh;        // [64][256]  (later reused as uS [256][64])
  __bf16* aggS = sm + 2 * Dd * Hh;    // [256][64]

  const int b    = blockIdx.x;
  const int t    = threadIdx.x;
  const int lane = t & 31;
  const int wv   = t >> 5;            // 8 waves
  const int mrow = lane & 15;
  const int hi   = lane >> 4;
  const int koff  = hi ? 8 : 0;       // bf16 A/B fragment k-chunk base
  const int mbase = hi ? 8 : 0;       // C fragment row base

  // ---- embed: h[i,:] = x[b,i]*wn[i,:] + bn[i,:]   (thread t -> feature row t)
  {
    const float xv = x[b * Dd + t];
    const __bf16* wr = wn_bf + t * Hh;
    const __bf16* br = bn_bf + t * Hh;
    __bf16* hr = hS + t * Hh;
#pragma unroll 8
    for (int h = 0; h < Hh; ++h)
      hr[h] = (__bf16)(xv * (float)wr[h] + (float)br[h]);
  }
  __syncthreads();

  const int rt0 = wv * 2;             // each wave owns 2 row-tiles x 4 col-tiles

  for (int r = 0; r < Rr; ++r) {
    const __bf16* AWT = aggwT + r * Hh * Hh;       // [64][64]  (N x K)
    const __bf16* W1T = w1T   + r * Hh * H2;       // [64][128]
    const __bf16* W2T = w2T   + r * Hh * Hh;       // [64][64]

    // per-lane biases for the 4 column tiles of this stage (hoisted)
    float bia[4], bi1[4], bi2[4];
#pragma unroll
    for (int ct = 0; ct < 4; ++ct) {
      bia[ct] = agg_b[r * Hh + ct * 16 + mrow];
      bi1[ct] = b1[r * Hh + ct * 16 + mrow];
      bi2[ct] = b2[r * Hh + ct * 16 + mrow];
    }

    // ---- GEMM1: msg = hS @ agg_w + agg_b  ->  msgT (transposed, b128 stores)
    for (int rt = rt0; rt < rt0 + 2; ++rt) {
      const int d0 = rt * 16;
      const v16bf a0 = ld_frag(hS + (d0 + mrow) * Hh,      koff);
      const v16bf a1 = ld_frag(hS + (d0 + mrow) * Hh + 32, koff);
#pragma unroll
      for (int ct = 0; ct < 4; ++ct) {
        const int h0 = ct * 16;
        const v16bf b0 = ld_frag(AWT + (h0 + mrow) * Hh,      koff);
        const v16bf b1f = ld_frag(AWT + (h0 + mrow) * Hh + 32, koff);
        v8f c = {};
        c = wmma_bf16(a0, b0, c);
        c = wmma_bf16(a1, b1f, c);
        st_transposed(msgT + (h0 + mrow) * Dd + d0 + mbase, c, bia[ct]);
      }
    }
    __syncthreads();

    // ---- GEMM2: agg = W^T @ msg  (K=256)  ->  aggS row-major
    //      A-fragments (rows of W^T) hoisted: loaded once, reused for 4 tiles.
    for (int rt = rt0; rt < rt0 + 2; ++rt) {
      const int d0 = rt * 16;
      v16bf aW[8];
#pragma unroll
      for (int kd = 0; kd < 8; ++kd)
        aW[kd] = ld_frag(WT + (d0 + mrow) * Dd + kd * 32, koff);
#pragma unroll
      for (int ct = 0; ct < 4; ++ct) {
        const int h0 = ct * 16;
        v8f c = {};
#pragma unroll
        for (int kd = 0; kd < 8; ++kd) {
          const v16bf bb = ld_frag(msgT + (h0 + mrow) * Dd + kd * 32, koff);
          c = wmma_bf16(aW[kd], bb, c);
        }
        st_rowmajor(aggS + (d0 + mbase) * Hh + h0 + mrow, Hh, c, 0.0f, false);
      }
    }
    __syncthreads();

    // ---- GEMM3: u = relu([h|agg] @ w1 + b1)  ->  uS (reuse msgT buffer)
    __bf16* uS = msgT;                 // now row-major [256][64]
    for (int rt = rt0; rt < rt0 + 2; ++rt) {
      const int d0 = rt * 16;
      v16bf aZ[4];                     // A-fragments hoisted across column tiles
      aZ[0] = ld_frag(hS   + (d0 + mrow) * Hh,      koff);
      aZ[1] = ld_frag(hS   + (d0 + mrow) * Hh + 32, koff);
      aZ[2] = ld_frag(aggS + (d0 + mrow) * Hh,      koff);
      aZ[3] = ld_frag(aggS + (d0 + mrow) * Hh + 32, koff);
#pragma unroll
      for (int ct = 0; ct < 4; ++ct) {
        const int h0 = ct * 16;
        v8f c = {};
#pragma unroll
        for (int k = 0; k < 4; ++k) {  // z = [h | agg] split over K=128
          const v16bf bb = ld_frag(W1T + (h0 + mrow) * H2 + k * 32, koff);
          c = wmma_bf16(aZ[k], bb, c);
        }
        st_rowmajor(uS + (d0 + mbase) * Hh + h0 + mrow, Hh, c, bi1[ct], true);
      }
    }
    __syncthreads();

    // ---- GEMM4: h' = uS @ w2 + b2  ->  hS (row-major)
    for (int rt = rt0; rt < rt0 + 2; ++rt) {
      const int d0 = rt * 16;
      v16bf aU[2];
      aU[0] = ld_frag(uS + (d0 + mrow) * Hh,      koff);
      aU[1] = ld_frag(uS + (d0 + mrow) * Hh + 32, koff);
#pragma unroll
      for (int ct = 0; ct < 4; ++ct) {
        const int h0 = ct * 16;
        v8f c = {};
#pragma unroll
        for (int k = 0; k < 2; ++k) {
          const v16bf bb = ld_frag(W2T + (h0 + mrow) * Hh + k * 32, koff);
          c = wmma_bf16(aU[k], bb, c);
        }
        st_rowmajor(hS + (d0 + mbase) * Hh + h0 + mrow, Hh, c, bi2[ct], false);
      }
    }
    __syncthreads();
  }

  // ---- spill final h (bf16) for the predictor GEMM: 32 KB, b128 copies
  {
    const int4* src = (const int4*)hS;
    int4* dst = (int4*)(hflat + (size_t)b * FLAT);
#pragma unroll
    for (int i = 0; i < 8; ++i) dst[t + i * 256] = src[t + i * 256];
  }
}

// ---------------------------------------------------------------------------
// Kernel 3: predictor.  Block = 16 batches; wave w owns q-columns [16w,16w+16).
// q = relu(hflat @ p1 + p1_b)  (WMMA, K=16384) ;  logits = q @ p2 + p2_b (VALU)
// ---------------------------------------------------------------------------
__global__ __launch_bounds__(256) void k_pred(
    const __bf16* __restrict__ hflat, const __bf16* __restrict__ p1T,
    const float* __restrict__ p1_b, const float* __restrict__ p2w,
    const float* __restrict__ p2b, float* __restrict__ out) {
  __shared__ float qS[16][H2];        // 8 KB
  const int t    = threadIdx.x;
  const int lane = t & 31;
  const int wv   = t >> 5;
  const int mrow = lane & 15;
  const int hi   = lane >> 4;
  const int koff  = hi ? 8 : 0;
  const int mbase = hi ? 8 : 0;
  const int mb = blockIdx.x * 16;     // batch tile
  const int n0 = wv * 16;             // q column tile

  const __bf16* arow = hflat + (size_t)(mb + mrow) * FLAT;
  const __bf16* brow = p1T   + (size_t)(n0 + mrow) * FLAT;

  v8f c = {};
  for (int ks = 0; ks < FLAT / 32; ++ks) {   // 512 k-steps
    Frag16 a, bf;
    a.q[0]  = *(const int4*)(arow + ks * 32 + koff);
    a.q[1]  = *(const int4*)(arow + ks * 32 + koff + 16);
    bf.q[0] = *(const int4*)(brow + ks * 32 + koff);
    bf.q[1] = *(const int4*)(brow + ks * 32 + koff + 16);
    c = wmma_bf16(a.v, bf.v, c);
  }
  const float bias = p1_b[n0 + mrow];
#pragma unroll
  for (int v = 0; v < 8; ++v)
    qS[mbase + v][n0 + mrow] = fmaxf(c[v] + bias, 0.0f);
  __syncthreads();

  if (t < 16 * Cc) {
    const int m = t / Cc, cc = t - m * Cc;
    float acc = p2b[cc];
#pragma unroll 8
    for (int k = 0; k < H2; ++k) acc += qS[m][k] * p2w[k * Cc + cc];
    out[(mb + m) * Cc + cc] = acc;
  }
}

// ---------------------------------------------------------------------------
// Host launcher
// ---------------------------------------------------------------------------
extern "C" void kernel_launch(void* const* d_in, const int* in_sizes, int n_in,
                              void* d_out, int out_size, void* d_ws, size_t ws_size,
                              hipStream_t stream) {
  const float* x        = (const float*)d_in[0];
  const float* wn       = (const float*)d_in[1];
  const float* bn       = (const float*)d_in[2];
  const float* A_raw    = (const float*)d_in[3];
  const float* priority = (const float*)d_in[4];
  const float* edge_l   = (const float*)d_in[5];
  const float* agg_w    = (const float*)d_in[6];
  const float* agg_b    = (const float*)d_in[7];
  const float* upd_w1   = (const float*)d_in[8];
  const float* upd_b1   = (const float*)d_in[9];
  const float* upd_w2   = (const float*)d_in[10];
  const float* upd_b2   = (const float*)d_in[11];
  const float* p1_w     = (const float*)d_in[12];
  const float* p1_b     = (const float*)d_in[13];
  const float* p2_w     = (const float*)d_in[14];
  const float* p2_b     = (const float*)d_in[15];
  float* out = (float*)d_out;

  // ---- carve bf16 workspace
  char* ws = (char*)d_ws;
  size_t off = 0;
  auto carve = [&](size_t bytes) { char* p = ws + off; off += (bytes + 255) & ~size_t(255); return p; };
  __bf16* WT_bf   = (__bf16*)carve(Dd * Dd * 2);            // W^T
  __bf16* wn_bf   = (__bf16*)carve(Dd * Hh * 2);
  __bf16* bn_bf   = (__bf16*)carve(Dd * Hh * 2);
  __bf16* aggwT   = (__bf16*)carve(Rr * Hh * Hh * 2);
  __bf16* w1T     = (__bf16*)carve(Rr * H2 * Hh * 2);
  __bf16* w2T     = (__bf16*)carve(Rr * Hh * Hh * 2);
  __bf16* p1T     = (__bf16*)carve((size_t)FLAT * H2 * 2);
  __bf16* hflat   = (__bf16*)carve((size_t)Bb * FLAT * 2);  // 128 MB
  (void)ws_size; (void)in_sizes; (void)n_in; (void)out_size;

  // ---- DAG matrix + weight conversion/transposition (all tiny vs main work)
  k_wgraph<<<1, 256, 0, stream>>>(A_raw, priority, edge_l, WT_bf);
  k_cvt<<<64, 256, 0, stream>>>(wn, wn_bf, Dd * Hh);
  k_cvt<<<64, 256, 0, stream>>>(bn, bn_bf, Dd * Hh);
  for (int r = 0; r < Rr; ++r) {
    k_tr<<<16, 256, 0, stream>>>(agg_w + r * Hh * Hh, aggwT + r * Hh * Hh, Hh, Hh);
    k_tr<<<32, 256, 0, stream>>>(upd_w1 + r * H2 * Hh, w1T + r * Hh * H2, H2, Hh);
    k_tr<<<16, 256, 0, stream>>>(upd_w2 + r * Hh * Hh, w2T + r * Hh * Hh, Hh, Hh);
  }
  k_tr<<<4096, 256, 0, stream>>>(p1_w, p1T, FLAT, H2);

  // ---- fused message-passing network (96 KB dynamic LDS per block)
  const int lds_bytes = 3 * Dd * Hh * 2;   // 98304
  hipFuncSetAttribute((const void*)k_main,
                      hipFuncAttributeMaxDynamicSharedMemorySize, lds_bytes);
  k_main<<<Bb, 256, lds_bytes, stream>>>(x, wn_bf, bn_bf, WT_bf,
                                         aggwT, agg_b, w1T, upd_b1, w2T, upd_b2,
                                         hflat);

  // ---- predictor
  k_pred<<<Bb / 16, 256, 0, stream>>>(hflat, p1T, p1_b, p2_w, p2_b, out);
}
